// JPmHC_40286793236987
// MI455X (gfx1250) — matrix operand order repaired and checked by reading, not compiled, not run on hardware
//
#include <hip/hip_runtime.h>
#include <hip/hip_bf16.h>
#include <cstddef>

// ---------------------------------------------------------------------------
// Shapes (from reference): B=4, S=4096, n=4, d=1024
//   tokens   = B*S = 16384
//   ND       = n*d = 4096          (GEMM K)
//   NOUT     = 3*n*n = 48          (GEMM N, = 3 tiles of 16)
// Per token the whole network collapses to a 4x4 mix matrix M applied to the
// 4 stream rows:  out[i,:] = sum_j M[i,j] * streams[j,:]
// ---------------------------------------------------------------------------

typedef __attribute__((ext_vector_type(16))) _Float16 v16h;
typedef __attribute__((ext_vector_type(8)))  float    v8f;
typedef __attribute__((ext_vector_type(4)))  float    v4f;   // b128, NT-store legal

#define ND    4096
#define NOUT  48
#define EPS   1e-5f
#define ALPHA 0.1f

// --- Kernel 0: wT[j][k] = proj_w[j][k] * norm_scale[k], converted to f16 ----
__global__ void prep_wT_kernel(const float* __restrict__ proj_w,
                               const float* __restrict__ norm_scale,
                               _Float16* __restrict__ wT, int total) {
    int idx = blockIdx.x * blockDim.x + threadIdx.x;
    if (idx >= total) return;
    int j = idx / ND;
    int k = idx - j * ND;
    wT[idx] = (_Float16)(proj_w[(size_t)j * ND + k] * norm_scale[k]);
}

// --- Kernel 1: fused projection (WMMA) + per-token 4x4 math + stream mix ----
__global__ __launch_bounds__(256) void fused_mix_kernel(
    const float*    __restrict__ streams,    // [tokens][4][1024] fp32
    const _Float16* __restrict__ wT,         // [48][4096] f16 (scale baked in)
    const float*    __restrict__ proj_b,     // [48]
    const float*    __restrict__ out_scale,  // [1]
    float*          __restrict__ out)        // [tokens][4][1024] fp32
{
    __shared__ float raw_s[8][16][NOUT];  // per-wave raw logits (post-norm)
    __shared__ float m_s[8][16][16];      // per-wave collapsed 4x4 matrices

    const int lane = threadIdx.x & 31;
    const int w    = threadIdx.x >> 5;           // wave in block (0..7)
    const int tile = blockIdx.x * 8 + w;         // 16-token tile id
    const int tok_base = tile * 16;
    const int m16 = lane & 15;                   // row of A / column of C
    const int hi  = lane >> 4;                   // wave half

    // -------- Phase 1: GEMM raw_unnorm[16][48] = x_f16 @ wT, plus sum(x^2) --
    const float* rowp = streams + (size_t)(tok_base + m16) * ND;

    v8f c0 = {}, c1 = {}, c2 = {};
    float sq0 = 0.f, sq1 = 0.f, sq2 = 0.f, sq3 = 0.f;  // 4 independent chains

    for (int k0 = 0; k0 < ND; k0 += 32) {
        // A tile 16x32 f16: lane (m16,hi) holds K = k0+hi*8+{0..7} and
        //                                     K = k0+16+hi*8+{0..7}
        const v4f* pa = reinterpret_cast<const v4f*>(rowp + k0 + hi * 8);
        v4f a0 = pa[0], a1 = pa[1];
        const v4f* pa2 = reinterpret_cast<const v4f*>(rowp + k0 + 16 + hi * 8);
        v4f a2 = pa2[0], a3 = pa2[1];

        sq0 = fmaf(a0.x, a0.x, sq0); sq1 = fmaf(a0.y, a0.y, sq1);
        sq2 = fmaf(a0.z, a0.z, sq2); sq3 = fmaf(a0.w, a0.w, sq3);
        sq0 = fmaf(a1.x, a1.x, sq0); sq1 = fmaf(a1.y, a1.y, sq1);
        sq2 = fmaf(a1.z, a1.z, sq2); sq3 = fmaf(a1.w, a1.w, sq3);
        sq0 = fmaf(a2.x, a2.x, sq0); sq1 = fmaf(a2.y, a2.y, sq1);
        sq2 = fmaf(a2.z, a2.z, sq2); sq3 = fmaf(a2.w, a2.w, sq3);
        sq0 = fmaf(a3.x, a3.x, sq0); sq1 = fmaf(a3.y, a3.y, sq1);
        sq2 = fmaf(a3.z, a3.z, sq2); sq3 = fmaf(a3.w, a3.w, sq3);

        v16h A;
        A[0]=(_Float16)a0.x; A[1]=(_Float16)a0.y; A[2]=(_Float16)a0.z; A[3]=(_Float16)a0.w;
        A[4]=(_Float16)a1.x; A[5]=(_Float16)a1.y; A[6]=(_Float16)a1.z; A[7]=(_Float16)a1.w;
        A[8]=(_Float16)a2.x; A[9]=(_Float16)a2.y; A[10]=(_Float16)a2.z; A[11]=(_Float16)a2.w;
        A[12]=(_Float16)a3.x; A[13]=(_Float16)a3.y; A[14]=(_Float16)a3.z; A[15]=(_Float16)a3.w;

        // B tiles 32x16 f16: lane supplies column N = m16 (+16*tileN),
        // K = k0+hi*8+{0..7} and K = k0+16+hi*8+{0..7} (2x b128 loads).
        const int krow = k0 + hi * 8;
        const v4f* pw = reinterpret_cast<const v4f*>(wT + (size_t)m16 * ND + krow);
        union U { v4f f[2]; v16h h; };
        U b0; b0.f[0] = pw[0]; b0.f[1] = pw[2];
        U b1; b1.f[0] = pw[8192]; b1.f[1] = pw[8194];
        U b2; b2.f[0] = pw[16384]; b2.f[1] = pw[16386];

        c0 = __builtin_amdgcn_wmma_f32_16x16x32_f16(false, A, false, b0.h, (short)0, c0, false, false);
        c1 = __builtin_amdgcn_wmma_f32_16x16x32_f16(false, A, false, b1.h, (short)0, c1, false, false);
        c2 = __builtin_amdgcn_wmma_f32_16x16x32_f16(false, A, false, b2.h, (short)0, c2, false, false);
    }

    // Full-row sum of squares: halves of the wave covered disjoint K subsets.
    float sumsq = (sq0 + sq1) + (sq2 + sq3);
    sumsq += __shfl_xor(sumsq, 16);
    const float inv = rsqrtf(sumsq * (1.0f / (float)ND) + EPS);  // 1/rms

    // C layout: VGPR r -> token (r + hi*8), column N = m16 (+16*tileN).
    #pragma unroll
    for (int r = 0; r < 8; ++r) {
        const int tr = r + hi * 8;
        const float invT = __shfl(inv, tr);      // lane tr holds token tr's 1/rms
        raw_s[w][tr][m16 +  0] = c0[r] * invT + proj_b[ 0 + m16];
        raw_s[w][tr][m16 + 16] = c1[r] * invT + proj_b[16 + m16];
        raw_s[w][tr][m16 + 32] = c2[r] * invT + proj_b[32 + m16];
    }
    __syncthreads();

    // -------- Phase 2: per-token 4x4 math -> collapsed mix matrix M --------
    if (lane < 16) {
        const float* R = raw_s[w][lane];

        float Hpre[16], Hpost[16];
        // H_pre: softmax over rows of R[0..15]
        #pragma unroll
        for (int i = 0; i < 4; ++i) {
            float mx = R[i*4+0];
            #pragma unroll
            for (int j = 1; j < 4; ++j) mx = fmaxf(mx, R[i*4+j]);
            float e[4], s = 0.f;
            #pragma unroll
            for (int j = 0; j < 4; ++j) { e[j] = __expf(R[i*4+j] - mx); s += e[j]; }
            const float is = 1.f / s;
            #pragma unroll
            for (int j = 0; j < 4; ++j) Hpre[i*4+j] = e[j] * is;
        }
        // H_post: softmax over columns of R[16..31]
        #pragma unroll
        for (int j = 0; j < 4; ++j) {
            float mx = R[16 + j];
            #pragma unroll
            for (int i = 1; i < 4; ++i) mx = fmaxf(mx, R[16 + i*4 + j]);
            float e[4], s = 0.f;
            #pragma unroll
            for (int i = 0; i < 4; ++i) { e[i] = __expf(R[16 + i*4 + j] - mx); s += e[i]; }
            const float is = 1.f / s;
            #pragma unroll
            for (int i = 0; i < 4; ++i) Hpost[i*4+j] = e[i] * is;
        }
        // Cayley fixed point: W = h2 - h2^T; Y = I + a*W; 2x Y = I + (a/2)*W*(I+Y)
        float Wm[16], Y[16];
        #pragma unroll
        for (int i = 0; i < 4; ++i)
            #pragma unroll
            for (int j = 0; j < 4; ++j)
                Wm[i*4+j] = R[32 + i*4 + j] - R[32 + j*4 + i];
        #pragma unroll
        for (int i = 0; i < 4; ++i)
            #pragma unroll
            for (int j = 0; j < 4; ++j)
                Y[i*4+j] = (i == j ? 1.f : 0.f) + ALPHA * Wm[i*4+j];
        #pragma unroll
        for (int s = 0; s < 2; ++s) {
            float T[16], Yn[16];
            #pragma unroll
            for (int i = 0; i < 4; ++i)
                #pragma unroll
                for (int j = 0; j < 4; ++j)
                    T[i*4+j] = (i == j ? 1.f : 0.f) + Y[i*4+j];
            #pragma unroll
            for (int i = 0; i < 4; ++i)
                #pragma unroll
                for (int j = 0; j < 4; ++j) {
                    float acc = (i == j ? 1.f : 0.f);
                    #pragma unroll
                    for (int k = 0; k < 4; ++k)
                        acc += (ALPHA * 0.5f) * Wm[i*4+k] * T[k*4+j];
                    Yn[i*4+j] = acc;
                }
            #pragma unroll
            for (int e = 0; e < 16; ++e) Y[e] = Yn[e];
        }
        // Collapse: M[i][j] = Y[i][j] + out_scale * rowsum(Hpost)[i] * colmean(Hpre)[j]
        float cj[4], ri[4];
        #pragma unroll
        for (int j = 0; j < 4; ++j)
            cj[j] = 0.25f * (Hpre[j] + Hpre[4+j] + Hpre[8+j] + Hpre[12+j]);
        #pragma unroll
        for (int i = 0; i < 4; ++i)
            ri[i] = Hpost[i*4] + Hpost[i*4+1] + Hpost[i*4+2] + Hpost[i*4+3];
        const float os = out_scale[0];
        #pragma unroll
        for (int i = 0; i < 4; ++i)
            #pragma unroll
            for (int j = 0; j < 4; ++j)
                m_s[w][lane][i*4+j] = Y[i*4+j] + os * ri[i] * cj[j];
    }
    __syncthreads();

    // -------- Phase 3: out[i,:] = sum_j M[i,j] * streams[j,:] --------------
    // Second read of streams is temporally adjacent to phase 1 -> L2 resident.
    // Output is write-once: non-temporal stores keep it from evicting the
    // streams lines we still want in L2.
    for (int t = 0; t < 16; ++t) {
        float M[16];
        #pragma unroll
        for (int e = 0; e < 16; ++e) M[e] = m_s[w][t][e];

        const size_t base = (size_t)(tok_base + t) * ND;
        const float* sp = streams + base;
        float*       op = out + base;

        for (int c0i = 0; c0i < 1024; c0i += 128) {
            const int col = c0i + lane * 4;
            const v4f s0 = *reinterpret_cast<const v4f*>(sp + 0*1024 + col);
            const v4f s1 = *reinterpret_cast<const v4f*>(sp + 1*1024 + col);
            const v4f s2 = *reinterpret_cast<const v4f*>(sp + 2*1024 + col);
            const v4f s3 = *reinterpret_cast<const v4f*>(sp + 3*1024 + col);
            #pragma unroll
            for (int i = 0; i < 4; ++i) {
                v4f o = M[i*4+0]*s0 + M[i*4+1]*s1 + M[i*4+2]*s2 + M[i*4+3]*s3;
                __builtin_nontemporal_store(o, reinterpret_cast<v4f*>(op + i*1024 + col));
            }
        }
    }
}

extern "C" void kernel_launch(void* const* d_in, const int* in_sizes, int n_in,
                              void* d_out, int out_size, void* d_ws, size_t ws_size,
                              hipStream_t stream) {
    const float* streams    = (const float*)d_in[0];
    const float* norm_scale = (const float*)d_in[1];
    const float* proj_w     = (const float*)d_in[2];
    const float* proj_b     = (const float*)d_in[3];
    const float* out_scale  = (const float*)d_in[4];
    float*       out        = (float*)d_out;
    _Float16*    wT         = (_Float16*)d_ws;   // [48][4096] f16 = 384 KB

    const int n_tokens = in_sizes[0] / ND;       // 16384

    // Kernel 0: transposed, scale-baked f16 weights.
    {
        const int total  = ND * NOUT;
        const int blocks = (total + 255) / 256;
        prep_wT_kernel<<<blocks, 256, 0, stream>>>(proj_w, norm_scale, wT, total);
    }
    // Kernel 1: fused WMMA projection + 4x4 collapse + stream mix.
    {
        const int tiles  = n_tokens / 16;        // 1024
        const int blocks = tiles / 8;            // 128 blocks of 8 waves
        fused_mix_kernel<<<blocks, 256, 0, stream>>>(streams, wT, proj_b,
                                                     out_scale, out);
    }
}
